// RecurrentFFNet_14731737825659
// MI455X (gfx1250) — matrix operand order repaired and checked by reading, not compiled, not run on hardware
//
#include <hip/hip_runtime.h>
#include <stdint.h>

typedef __attribute__((ext_vector_type(16))) __bf16 v16bf;
typedef __attribute__((ext_vector_type(8)))  __bf16 v8bf;
typedef __attribute__((ext_vector_type(8)))  float  v8f;

#define B_DIM   8192
#define FEAT    2000
#define SEG_PAD 2048                  // 2000 padded to multiple of BK
#define K_PAD   (3*SEG_PAD)          // 6144
#define N_PAD   2048                 // SIZE padded to multiple of BN
#define EPS_    1e-8f
#define DAMP    0.7f

#define BM 128
#define BN 128
#define BK 64
#define KSTEPS (K_PAD / BK)          // 96

#if defined(__has_builtin)
#if __has_builtin(__builtin_amdgcn_global_load_async_to_lds_b128) && \
    __has_builtin(__builtin_amdgcn_s_wait_asynccnt)
#define HAVE_ASYNC 1
#endif
#endif
#ifndef HAVE_ASYNC
#define HAVE_ASYNC 0
#endif

#if HAVE_ASYNC
typedef int v4i_vec __attribute__((vector_size(16)));
typedef __attribute__((address_space(1))) v4i_vec gv4i;   // global int4
typedef __attribute__((address_space(3))) v4i_vec lv4i;   // LDS int4
// one 16-byte async copy, global -> LDS (per-lane addresses, ASYNCcnt-tracked)
static __device__ __forceinline__ void async_cp16(const __bf16* g, __bf16* l) {
  __builtin_amdgcn_global_load_async_to_lds_b128(
      (gv4i*)(v4i_vec*)g, (lv4i*)(v4i_vec*)l, 0, 0);
}
#else
static __device__ __forceinline__ void async_cp16(const __bf16* g, __bf16* l) {
  *(uint4*)l = *(const uint4*)g;   // synchronous fallback
}
#endif

// ---------------- Pass 1a: per-row standardize (ddof=1) + bf16 cast into A_cat ----
__global__ __launch_bounds__(256) void standardize_cast_kernel(
    const float* __restrict__ prev, const float* __restrict__ nxt,
    const float* __restrict__ self_, __bf16* __restrict__ Acat)
{
  const int row = blockIdx.x;
  const int seg = blockIdx.y;                 // 0=prev<->fW, 1=next<->bW, 2=self<->lW
  const float* in = (seg == 0) ? prev : ((seg == 1) ? nxt : self_);
  const float* r  = in + (size_t)row * FEAT;

  __shared__ float s_sum[256], s_sq[256];
  const int tid = threadIdx.x;
  float sum = 0.f, sq = 0.f;
  for (int j = tid; j < FEAT; j += 256) { float v = r[j]; sum += v; sq += v * v; }
  s_sum[tid] = sum; s_sq[tid] = sq;
  __syncthreads();
  for (int off = 128; off > 0; off >>= 1) {
    if (tid < off) { s_sum[tid] += s_sum[tid + off]; s_sq[tid] += s_sq[tid + off]; }
    __syncthreads();
  }
  const float tot  = s_sum[0], tsq = s_sq[0];
  const float mean = tot / (float)FEAT;
  const float var  = fmaxf((tsq - tot * tot / (float)FEAT) / (float)(FEAT - 1), 0.f);
  const float inv  = 1.0f / (sqrtf(var) + EPS_);

  __bf16* dst = Acat + (size_t)row * K_PAD + (size_t)seg * SEG_PAD;
  for (int j = tid; j < SEG_PAD; j += 256) {
    float v = (j < FEAT) ? (r[j] - mean) * inv : 0.f;
    dst[j] = (__bf16)v;
  }
}

// ---------------- Pass 1b: pack [fW|bW|lW] -> bf16 W_cat (zero padded) ------------
__global__ __launch_bounds__(256) void pack_weights_kernel(
    const float* __restrict__ fW, const float* __restrict__ bW,
    const float* __restrict__ lW, __bf16* __restrict__ Wcat)
{
  size_t idx = (size_t)blockIdx.x * 256 + threadIdx.x;
  const size_t total = (size_t)N_PAD * K_PAD;
  if (idx >= total) return;
  const int s   = (int)(idx / K_PAD);
  const int k   = (int)(idx % K_PAD);
  const int seg = k / SEG_PAD;
  const int kk  = k - seg * SEG_PAD;
  float v = 0.f;
  if (s < FEAT && kk < FEAT) {
    const float* W = (seg == 0) ? fW : ((seg == 1) ? bW : lW);
    v = W[(size_t)s * FEAT + kk];
  }
  Wcat[idx] = (__bf16)v;
}

__global__ __launch_bounds__(256) void bias_sum_kernel(
    const float* __restrict__ fb, const float* __restrict__ bb,
    const float* __restrict__ lb, float* __restrict__ bias)
{
  int s = blockIdx.x * 256 + threadIdx.x;
  if (s < FEAT) bias[s] = fb[s] + bb[s] + lb[s];
}

// ---------------- Pass 2: fused GEMM (8192 x 2000 x 6144) bf16 WMMA, async DB -----
__global__ __launch_bounds__(256) void gemm_wmma_kernel(
    const __bf16* __restrict__ Acat, const __bf16* __restrict__ Wcat,
    const float* __restrict__ bias,  const float* __restrict__ self_act,
    float* __restrict__ out)
{
  __shared__ __align__(32) __bf16 sA[2][BM * BK];   // 2 x 16 KB
  __shared__ __align__(32) __bf16 sB[2][BN * BK];   // 2 x 16 KB

  const int tid  = threadIdx.x;
  const int lane = tid & 31;
  const int wave = tid >> 5;          // 0..7
  const int wm   = wave & 1;          // 2 waves along M (64 rows each)
  const int wn   = wave >> 1;         // 4 waves along N (32 cols each)
  const int blockN = blockIdx.x * BN;
  const int blockM = blockIdx.y * BM;

  v8f acc[4][2] = {};                 // 4 M-tiles x 2 N-tiles of 16x16 f32

  // staging map: thread -> (row 0..127, 32-half chunk 0/1) = 64 B per tile
  const int lrow = tid >> 1;
  const int lcol = (tid & 1) * 32;

  // fragment addressing: contiguous k order (k = 16*(lane/16) + s), same for A and B
  const int l16  = lane & 15;
  const int khiA = (lane >> 4) * 16;  // A frag: lanes 16-31 hold K=16..31
  const int khiB = (lane >> 4) * 16;  // B frag: lanes 16-31 hold K=16..31

  const __bf16* gA = Acat + (size_t)(blockM + lrow) * K_PAD + lcol;
  const __bf16* gB = Wcat + (size_t)(blockN + lrow) * K_PAD + lcol;
  __bf16* lA0 = &sA[0][lrow * BK + lcol];
  __bf16* lB0 = &sB[0][lrow * BK + lcol];
  const int bufStrideA = BM * BK;     // halves between ping-pong buffers
  const int bufStrideB = BN * BK;

  // stage(step -> buffer): 4 async b128 per tile per thread (64 B)
  auto stage = [&](int buf, int k0) {
    const __bf16* ga = gA + k0;
    const __bf16* gb = gB + k0;
    __bf16* la = lA0 + buf * bufStrideA;
    __bf16* lb = lB0 + buf * bufStrideB;
#pragma unroll
    for (int c = 0; c < 4; ++c) {
      async_cp16(ga + c * 8, la + c * 8);   // 8 halves = 16 B
    }
#pragma unroll
    for (int c = 0; c < 4; ++c) {
      async_cp16(gb + c * 8, lb + c * 8);
    }
  };

  stage(0, 0);

  for (int step = 0; step < KSTEPS; ++step) {
    const int cur  = step & 1;
    const bool more = (step + 1 < KSTEPS);
    if (more) stage(cur ^ 1, (step + 1) * BK);
#if HAVE_ASYNC
    if (more) __builtin_amdgcn_s_wait_asynccnt(8);   // cur's 8 copies done (in-order)
    else      __builtin_amdgcn_s_wait_asynccnt(0);
#endif
    __syncthreads();                                  // cur buffer visible to all waves

    const __bf16* cA = &sA[cur][0];
    const __bf16* cB = &sB[cur][0];
#pragma unroll
    for (int ks = 0; ks < 2; ++ks) {                  // two K=32 sub-steps
      const int kofs = ks * 32;
      v16bf bfrag[2];
#pragma unroll
      for (int j = 0; j < 2; ++j) {
        const int nrow = wn * 32 + j * 16 + l16;
        bfrag[j] = *(const v16bf*)(cB + nrow * BK + kofs + khiB);
      }
#pragma unroll
      for (int i = 0; i < 4; ++i) {
        const int mrow = wm * 64 + i * 16 + l16;
        v8bf lo = *(const v8bf*)(cA + mrow * BK + kofs + khiA);
        v8bf hi = *(const v8bf*)(cA + mrow * BK + kofs + 8 + khiA);
        v16bf afrag = __builtin_shufflevector(lo, hi,
            0, 1, 2, 3, 4, 5, 6, 7, 8, 9, 10, 11, 12, 13, 14, 15);
#pragma unroll
        for (int j = 0; j < 2; ++j) {
          acc[i][j] = __builtin_amdgcn_wmma_f32_16x16x32_bf16(
              false, afrag, false, bfrag[j], (short)0, acc[i][j], false, false);
        }
      }
    }
    __syncthreads();   // all reads of cur done before it is overwritten next step
  }

  // epilogue: + bias, relu, damped recurrent blend
  const int mhalf = (lane >> 4) * 8;  // C/D: VGPR i -> M=i (lanes 0-15) / M=i+8
#pragma unroll
  for (int i = 0; i < 4; ++i) {
#pragma unroll
    for (int j = 0; j < 2; ++j) {
      const int n = blockN + wn * 32 + j * 16 + l16;
      if (n < FEAT) {
        const float bv = bias[n];
#pragma unroll
        for (int e = 0; e < 8; ++e) {
          const int m = blockM + wm * 64 + i * 16 + e + mhalf;
          const float pre  = acc[i][j][e] + bv;
          const float relu = fmaxf(pre, 0.f);
          const size_t o = (size_t)m * FEAT + n;
          out[o] = DAMP * relu + (1.0f - DAMP) * self_act[o];
        }
      }
    }
  }
}

extern "C" void kernel_launch(void* const* d_in, const int* in_sizes, int n_in,
                              void* d_out, int out_size, void* d_ws, size_t ws_size,
                              hipStream_t stream)
{
  (void)in_sizes; (void)n_in; (void)out_size; (void)ws_size;
  const float* prev  = (const float*)d_in[0];
  const float* self_ = (const float*)d_in[1];
  const float* nxt   = (const float*)d_in[2];
  const float* fW    = (const float*)d_in[3];
  const float* fb    = (const float*)d_in[4];
  const float* bW    = (const float*)d_in[5];
  const float* bb    = (const float*)d_in[6];
  const float* lW    = (const float*)d_in[7];
  const float* lb    = (const float*)d_in[8];
  float* out = (float*)d_out;

  char* ws = (char*)d_ws;
  const size_t Abytes = (size_t)B_DIM * K_PAD * sizeof(__bf16);   // ~100.7 MB
  const size_t Wbytes = (size_t)N_PAD * K_PAD * sizeof(__bf16);   // ~25.2 MB
  __bf16* Acat = (__bf16*)ws;
  __bf16* Wcat = (__bf16*)(ws + Abytes);
  float*  bias = (float*)(ws + Abytes + Wbytes);

  standardize_cast_kernel<<<dim3(B_DIM, 3), 256, 0, stream>>>(prev, nxt, self_, Acat);

  const size_t wtot = (size_t)N_PAD * K_PAD;
  pack_weights_kernel<<<(unsigned)((wtot + 255) / 256), 256, 0, stream>>>(fW, bW, lW, Wcat);
  bias_sum_kernel<<<(FEAT + 255) / 256, 256, 0, stream>>>(fb, bb, lb, bias);

  gemm_wmma_kernel<<<dim3(N_PAD / BN, B_DIM / BM), 256, 0, stream>>>(
      Acat, Wcat, bias, self_, out);
}
